// MultiHeadAttention_44513041056016
// MI455X (gfx1250) — matrix operand order repaired and checked by reading, not compiled, not run on hardware
//
#include <hip/hip_runtime.h>
#include <hip/hip_bf16.h>

// ---------------------------------------------------------------------------
// MHA forward for MI455X (gfx1250, wave32, WMMA f32_16x16x32_f16).
// B=2 T=2048 C=1024 H=16 DH=64.  Compute-bound (~69 GFLOP over ~50MB, and the
// whole working set fits the 192MB L2) -> inner loops are v_wmma + b128 loads;
// A-tiles staged to LDS with the gfx1250 async data mover (ASYNCcnt path).
// ---------------------------------------------------------------------------

typedef __attribute__((ext_vector_type(16))) _Float16 v16h;
typedef __attribute__((ext_vector_type(4)))  _Float16 h4;
typedef __attribute__((ext_vector_type(8)))  float    v8f;
typedef __attribute__((ext_vector_type(4)))  int      v4i;

#define BB 2
#define TT 2048
#define CC 1024
#define HH 16
#define DH 64

__device__ __forceinline__ v8f wmma_f16(v16h a, v16h b, v8f c) {
  // 8 args: (neg_a, A, neg_b, B, c_mod, C, reuse_a, reuse_b)
  return __builtin_amdgcn_wmma_f32_16x16x32_f16(false, a, false, b, (short)0, c,
                                                false, false);
}

// ---- gfx1250 async LDS staging (ASYNCcnt path), guarded by __has_builtin ---
#if __has_builtin(__builtin_amdgcn_global_load_async_to_lds_b128) && \
    __has_builtin(__builtin_amdgcn_s_wait_asynccnt)
#define HAVE_ASYNC_LDS 1
#endif

__device__ __forceinline__ void async_copy16(void* lds, const void* gsrc) {
#ifdef HAVE_ASYNC_LDS
  // flat LDS addresses carry the LDS byte offset in bits [31:0] (ISA 10.2),
  // so int->addrspace(3) pointer materializes the correct DS address.
  __builtin_amdgcn_global_load_async_to_lds_b128(
      (__attribute__((address_space(1))) v4i*)(uintptr_t)gsrc,
      (__attribute__((address_space(3))) v4i*)(unsigned)(uintptr_t)lds,
      0, 0);
#else
  *(uint4*)lds = *(const uint4*)gsrc;
#endif
}
__device__ __forceinline__ void async_join() {
#ifdef HAVE_ASYNC_LDS
  __builtin_amdgcn_s_wait_asynccnt(0);
#endif
}

// ---- fragment builders, contiguous k order (k = 16*(lane>>4) + e) for A and B --------------

// A matrix 16x32 f16 (row-major src, leading dim ld):
// lane&15 = row; lanes 0-15 hold K 0..15, lanes 16-31 K 16..31.
// Per lane: two contiguous 8-half chunks -> 2x (ds_)load_b128.
__device__ __forceinline__ v16h load_a_frag_f16(const _Float16* __restrict__ src,
                                                int ld) {
  const int lane = threadIdx.x & 31;
  const int row  = lane & 15;
  const int hi   = lane >> 4;
  v16h a;
#pragma unroll
  for (int e = 0; e < 16; ++e) {
    const int vg = e >> 1, h = e & 1;
    const int k = hi * 16 + e;
    a[e] = src[row * ld + k];
  }
  return a;
}

// B matrix 32x16 where logical B[k][n] = src[n*ld + k] (src holds the 16 "n"
// rows with K contiguous).  Per lane: 16 contiguous halves -> 2x b128.
// Used for Q*K^T (src = K[s,d] rows) and P*V (src = V^T[d,t] rows).
__device__ __forceinline__ v16h load_bT_frag_f16(const _Float16* __restrict__ src,
                                                 int ld) {
  const int lane = threadIdx.x & 31;
  const int n  = lane & 15;
  const int k0 = (lane >> 4) * 16;
  v16h b;
#pragma unroll
  for (int e = 0; e < 16; ++e) b[e] = src[(size_t)n * ld + k0 + e];
  return b;
}

// half-wave (16-lane) row reductions for the f32 C-layout (row r lives across
// lanes 0-15 of VGPR r for r<8, lanes 16-31 of VGPR r-8 for r>=8).
__device__ __forceinline__ float rowmax16(float x) {
#pragma unroll
  for (int m = 8; m >= 1; m >>= 1) x = fmaxf(x, __shfl_xor(x, m, 16));
  return x;
}
__device__ __forceinline__ float rowsum16(float x) {
#pragma unroll
  for (int m = 8; m >= 1; m >>= 1) x += __shfl_xor(x, m, 16);
  return x;
}

// ---------------------------------------------------------------------------
// x [B*T, C] f32 -> f16, one-time (removes cvt work + halves bytes for the
// 48x L2-resident re-reads by the QKV GEMM).
// ---------------------------------------------------------------------------
__global__ void convert_x_kernel(const float* __restrict__ x,
                                 _Float16* __restrict__ xh, int n4) {
  const int i = blockIdx.x * blockDim.x + threadIdx.x;
  if (i >= n4) return;
  const float4 f = ((const float4*)x)[i];
  h4 hv = {(_Float16)f.x, (_Float16)f.y, (_Float16)f.z, (_Float16)f.w};
  ((h4*)xh)[i] = hv;
}

// ---------------------------------------------------------------------------
// Weight pre-pack: f32 [K x N] row-major (nmat stacked matrices) -> f16
// B-fragments, one v16h per (tile, lane); tile = n16*(K/32) + k32.
// Makes every GEMM B-frag a single 32-byte load (2x global_load_b128).
// ---------------------------------------------------------------------------
__global__ void pack_b_frags(const float* __restrict__ src,
                             v16h* __restrict__ dst,
                             int K, int N, int total, size_t matStride) {
  const int gid = blockIdx.x * blockDim.x + threadIdx.x;
  if (gid >= total) return;
  const int lane = gid & 31;
  const int tile = gid >> 5;
  const int tilesPerMat = (K >> 5) * (N >> 4);
  const int mat = tile / tilesPerMat;
  const int t   = tile - mat * tilesPerMat;
  const int n16 = t / (K >> 5);
  const int k32 = t - n16 * (K >> 5);
  const float* s = src + (size_t)mat * matStride;
  const int n = lane & 15, khalf = (lane >> 4) * 16;
  v16h b;
#pragma unroll
  for (int e = 0; e < 16; ++e)
    b[e] = (_Float16)s[(size_t)(k32 * 32 + khalf + e) * N + n16 * 16 + n];
  dst[gid] = b;
}

// ---------------------------------------------------------------------------
// Kernel 1: q/k = xh @ W -> f16 [B,H,T,DH];  v -> transposed f16 [B,H,DH,T].
// grid (B*T/64, H, 3), block 128: 64x64 output tile, wave w = rows 16w..16w+15,
// 4 WMMAs per wave per 32-K step; A staged async to LDS, B from packed weights.
// ---------------------------------------------------------------------------
__global__ void mha_qkv_kernel(const _Float16* __restrict__ xh,
                               const v16h* __restrict__ pkw,
                               _Float16* __restrict__ qb,
                               _Float16* __restrict__ kb,
                               _Float16* __restrict__ vt) {
  const int rowTile = blockIdx.x;            // 64-row tiles over B*T
  const int h       = blockIdx.y;
  const int which   = blockIdx.z;            // 0=q 1=k 2=v

  __shared__ __align__(16) _Float16 sA[64 * 32];
  const int tid  = threadIdx.x;
  const int lane = tid & 31, wave = tid >> 5;
  const int colLane = lane & 15, hi = lane >> 4;

  const _Float16* xrow = xh + (size_t)rowTile * 64 * CC;
  const v16h* wbase = pkw + (size_t)(which * HH + h) * (128 * 32);

  v8f acc[4] = {};
  for (int kc = 0; kc < CC; kc += 32) {
    // stage 64x32 f16 X tile: 256 x 16B chunks via the async data mover
#pragma unroll
    for (int i = tid; i < 256; i += 128) {
      const int r = i >> 2, c8 = (i & 3) * 8;
      async_copy16(sA + r * 32 + c8, xrow + (size_t)r * CC + kc + c8);
    }
    async_join();
    __syncthreads();
    const v16h a = load_a_frag_f16(sA + wave * 16 * 32, 32);
    const int kt = kc >> 5;
#pragma unroll
    for (int n = 0; n < 4; ++n) {
      const v16h b = wbase[(size_t)(n * 32 + kt) * 32 + lane];
      acc[n] = wmma_f16(a, b, acc[n]);
    }
    __syncthreads();
  }

#pragma unroll
  for (int n = 0; n < 4; ++n)
#pragma unroll
    for (int i = 0; i < 8; ++i) {
      const int g = rowTile * 64 + wave * 16 + i + hi * 8;  // row in [0,B*T)
      const int bidx = g >> 11, t = g & (TT - 1);
      const int d = n * 16 + colLane;
      const _Float16 val = (_Float16)acc[n][i];
      if (which == 2)        // V stored transposed for contiguous PV B-frags
        vt[((size_t)(bidx * HH + h) * DH + d) * TT + t] = val;
      else {
        _Float16* dst = (which == 0) ? qb : kb;
        dst[((size_t)(bidx * HH + h) * TT + t) * DH + d] = val;
      }
    }
}

// ---------------------------------------------------------------------------
// Kernel 2: causal flash attention.  grid (T/64, B*H), block 128.
// Each wave independently owns 16 query rows and the full DH=64 output:
// no cross-wave redundancy, no barriers.  32 keys / iteration:
// 4 score WMMAs + 4 PV WMMAs per wave.
// ---------------------------------------------------------------------------
__global__ void mha_attn_kernel(const _Float16* __restrict__ q,
                                const _Float16* __restrict__ k,
                                const _Float16* __restrict__ vt,
                                _Float16* __restrict__ att) {
  const int qt64 = blockIdx.x;
  const int bh   = blockIdx.y;
  const int b = bh >> 4, h = bh & 15;

  const int tid  = threadIdx.x;
  const int lane = tid & 31, wave = tid >> 5;
  const int colLane = lane & 15, hi = lane >> 4;
  const int rbase = qt64 * 64 + wave * 16;

  const _Float16* qp = q  + ((size_t)bh * TT + rbase) * DH;
  const _Float16* kp = k  + (size_t)bh * TT * DH;
  const _Float16* vp = vt + (size_t)bh * DH * TT;

  __shared__ __align__(16) _Float16 sP[4][16 * 32];  // per-wave P relayout
  _Float16* sPw = sP[wave];

  const v16h qa0 = load_a_frag_f16(qp, DH);        // d = 0..31
  const v16h qa1 = load_a_frag_f16(qp + 32, DH);   // d = 32..63

  v8f acc[4] = {};
  float m[8], l[8];
#pragma unroll
  for (int i = 0; i < 8; ++i) { m[i] = -3.0e38f; l[i] = 0.0f; }

  const float scale = 0.125f;                      // 1/sqrt(64)

  for (int kbase = 0; kbase <= rbase + 15; kbase += 32) {
    __builtin_prefetch(kp + (size_t)(kbase + 32) * DH, 0, 0);

    // scores for 32 keys: two 16x16 f32 tiles, 2 WMMAs each over DH=64
    v8f s0 = {}, s1 = {};
    s0 = wmma_f16(qa0, load_bT_frag_f16(kp + (size_t)kbase * DH, DH), s0);
    s0 = wmma_f16(qa1, load_bT_frag_f16(kp + (size_t)kbase * DH + 32, DH), s0);
    s1 = wmma_f16(qa0, load_bT_frag_f16(kp + (size_t)(kbase + 16) * DH, DH), s1);
    s1 = wmma_f16(qa1, load_bT_frag_f16(kp + (size_t)(kbase + 16) * DH + 32, DH), s1);

#pragma unroll
    for (int i = 0; i < 8; ++i) {
      const int rowg = rbase + i + hi * 8;
      float a0 = s0[i] * scale;
      float a1 = s1[i] * scale;
      a0 = (kbase + colLane      > rowg) ? -3.0e38f : a0;   // causal mask
      a1 = (kbase + 16 + colLane > rowg) ? -3.0e38f : a1;
      const float mc = rowmax16(fmaxf(a0, a1));
      const float mn = fmaxf(m[i], mc);
      const float corr = __expf(m[i] - mn);
      const float p0 = __expf(a0 - mn);
      const float p1 = __expf(a1 - mn);
      l[i] = l[i] * corr + rowsum16(p0 + p1);
      m[i] = mn;
      acc[0][i] *= corr; acc[1][i] *= corr;
      acc[2][i] *= corr; acc[3][i] *= corr;
      // park probabilities in LDS (row-major 16x32 f16) for A-frag reload
      sPw[(i + hi * 8) * 32 + colLane]      = (_Float16)p0;
      sPw[(i + hi * 8) * 32 + 16 + colLane] = (_Float16)p1;
    }

    const v16h pa = load_a_frag_f16(sPw, 32);            // P 16x32
#pragma unroll
    for (int n = 0; n < 4; ++n) {                        // V^T rows: contiguous
      const v16h vb = load_bT_frag_f16(vp + (size_t)(n * 16) * TT + kbase, TT);
      acc[n] = wmma_f16(pa, vb, acc[n]);
    }
  }

  // heads concatenated directly: att[b*T + t, h*64 + d]  (f16)
#pragma unroll
  for (int n = 0; n < 4; ++n)
#pragma unroll
    for (int i = 0; i < 8; ++i) {
      const size_t row = (size_t)b * TT + rbase + i + hi * 8;
      att[row * CC + h * DH + n * 16 + colLane] = (_Float16)(acc[n][i] / l[i]);
    }
}

// ---------------------------------------------------------------------------
// Kernel 3: out = att @ Wp + bp.  grid (B*T/64, C/64), block 128.
// Same 64x64 WMMA tile structure as kernel 1; packed Wp fragments.
// ---------------------------------------------------------------------------
__global__ void mha_proj_kernel(const _Float16* __restrict__ att,
                                const v16h* __restrict__ pkw,
                                const float* __restrict__ bp,
                                float* __restrict__ out) {
  const int rowTile = blockIdx.x;
  const int ncTile  = blockIdx.y;

  __shared__ __align__(16) _Float16 sA[64 * 32];
  const int tid  = threadIdx.x;
  const int lane = tid & 31, wave = tid >> 5;
  const int colLane = lane & 15, hi = lane >> 4;

  const _Float16* arow = att + (size_t)rowTile * 64 * CC;
  v8f acc[4] = {};
  for (int kc = 0; kc < CC; kc += 32) {
#pragma unroll
    for (int i = tid; i < 256; i += 128) {
      const int r = i >> 2, c8 = (i & 3) * 8;
      async_copy16(sA + r * 32 + c8, arow + (size_t)r * CC + kc + c8);
    }
    async_join();
    __syncthreads();
    const v16h a = load_a_frag_f16(sA + wave * 16 * 32, 32);
    const int kt = kc >> 5;
#pragma unroll
    for (int n = 0; n < 4; ++n) {
      const v16h b = pkw[(size_t)((ncTile * 4 + n) * 32 + kt) * 32 + lane];
      acc[n] = wmma_f16(a, b, acc[n]);
    }
    __syncthreads();
  }

#pragma unroll
  for (int n = 0; n < 4; ++n) {
    const int colg = ncTile * 64 + n * 16 + colLane;
    const float bias = bp[colg];
#pragma unroll
    for (int i = 0; i < 8; ++i) {
      const size_t row = (size_t)rowTile * 64 + wave * 16 + i + hi * 8;
      out[row * CC + colg] = acc[n][i] + bias;
    }
  }
}

// ---------------------------------------------------------------------------

extern "C" void kernel_launch(void* const* d_in, const int* in_sizes, int n_in,
                              void* d_out, int out_size, void* d_ws, size_t ws_size,
                              hipStream_t stream) {
  const float* x  = (const float*)d_in[0];   // [B,T,C]
  const float* Wq = (const float*)d_in[1];   // [H,C,DH]
  const float* Wk = (const float*)d_in[2];
  const float* Wv = (const float*)d_in[3];
  const float* Wp = (const float*)d_in[4];   // [C,C]
  const float* bp = (const float*)d_in[5];   // [C]
  float* out = (float*)d_out;                // [B,T,C]
  (void)in_sizes; (void)n_in; (void)out_size; (void)ws_size;

  const size_t xElems   = (size_t)BB * TT * CC;        // 4 Mi
  const size_t qkvElems = (size_t)BB * HH * TT * DH;   // 4 Mi f16
  _Float16* xhb  = (_Float16*)d_ws;
  _Float16* qbuf = xhb + xElems;
  _Float16* kbuf = qbuf + qkvElems;
  _Float16* vtb  = kbuf + qkvElems;            // V transposed [B,H,DH,T]
  _Float16* atb  = vtb + qkvElems;
  v16h* pkQKV = (v16h*)(atb + xElems);         // 3*16*128 tiles * 32 lanes
  v16h* pkWp  = pkQKV + (size_t)3 * HH * 128 * 32;

  // ---- one-time preps: x -> f16, weights -> packed B-fragments ----
  convert_x_kernel<<<(int)(xElems / 4 / 256), 256, 0, stream>>>(
      x, xhb, (int)(xElems / 4));
  const int qkvTotal = HH * 128 * 32;          // per projection: 65536 v16h
  pack_b_frags<<<qkvTotal / 128, 128, 0, stream>>>(
      Wq, pkQKV, CC, DH, qkvTotal, (size_t)CC * DH);
  pack_b_frags<<<qkvTotal / 128, 128, 0, stream>>>(
      Wk, pkQKV + qkvTotal, CC, DH, qkvTotal, (size_t)CC * DH);
  pack_b_frags<<<qkvTotal / 128, 128, 0, stream>>>(
      Wv, pkQKV + 2 * qkvTotal, CC, DH, qkvTotal, (size_t)CC * DH);
  const int wpTotal = (CC / 32) * (CC / 16) * 32;   // 65536 v16h
  pack_b_frags<<<wpTotal / 128, 128, 0, stream>>>(
      Wp, pkWp, CC, CC, wpTotal, (size_t)CC * CC);

  // 1) QKV projections (64x64 WMMA tiles; V written transposed)
  mha_qkv_kernel<<<dim3(BB * TT / 64, HH, 3), 128, 0, stream>>>(
      xhb, pkQKV, qbuf, kbuf, vtb);

  // 2) causal flash attention (per-wave 16 query rows, barrier-free)
  mha_attn_kernel<<<dim3(TT / 64, BB * HH), 128, 0, stream>>>(
      qbuf, kbuf, vtb, atb);

  // 3) output projection + bias
  mha_proj_kernel<<<dim3(BB * TT / 64, CC / 64), 128, 0, stream>>>(
      atb, pkWp, bp, out);
}
